// InteractionBlock_52080773431364
// MI455X (gfx1250) — matrix operand order repaired and checked by reading, not compiled, not run on hardware
//
#include <hip/hip_runtime.h>
#include <hip/hip_bf16.h>

// ---------------------------------------------------------------------------
// MI455X (gfx1250) fused equivariant interaction block.
//   K0: zero agg + stats          K1: weights -> permuted column-major bf16
//   K2: per-edge  [MLP(8->64->64->768) via v_wmma_f32_16x16x32_bf16]
//       + tensor product epilogue + atomic scatter to agg
//   K3: per-node gates + self-interaction + BN statistics
//   K4: batch-norm affine + residual
// ---------------------------------------------------------------------------

#define N_NODES   50000
#define N_EDGES   300000
#define DIM_IN    40
#define NTILE     (N_EDGES / 16)     // 18750 16-edge tiles
#define WAVES     8                  // waves (tiles) per block
#define ALPHA     0.2041241452319315f    // 1/sqrt(24)
#define INV_SQRT3 0.5773502691896258f
#define INV_SQRT8 0.35355339059327373f

typedef __bf16 bf16_t;
typedef __attribute__((ext_vector_type(16))) __bf16 v16bf;
typedef __attribute__((ext_vector_type(8)))  __bf16 v8bf;
typedef __attribute__((ext_vector_type(8)))  float  v8f;

// workspace layout (float indices)
#define OFF_AGG   0            // 50000*48 = 2,400,000
#define OFF_STATS 2400000      // 64
#define OFF_SV    2400064      // 50000*40 = 2,000,000
#define OFF_B3P   4400064      // 768
#define OFF_WBF16 4400832      // 53760 bf16 (= 26880 floats)

// Fast activations: v_exp_f32 + v_rcp_f32 instead of IEEE division
// (avoids v_div_scale/v_div_fmas Newton chains in the WMMA epilogue).
__device__ __forceinline__ float sigmoid_f(float x) {
  return __builtin_amdgcn_rcpf(1.0f + __expf(-x));
}
__device__ __forceinline__ float silu_f(float x) { return x * sigmoid_f(x); }

// Column permutation of W3: new column n -> original column.
// n in [0,384):   w_ss, chunk w holds u=0..15            -> orig = u*24 + w
// n in [384,576): w_vv, chunk t: cols<8 -> w=2t, else w=2t+1, u=col&7
// n in [576,704): w_sv, chunk w holds u=0..15            -> orig = 576 + u*8 + w
// n in [704,768): w_vs, chunk t: cols<8 -> w=2t, else w=2t+1, u=col&7
__device__ __forceinline__ int origcol(int n) {
  if (n < 384) { int w = n >> 4, u = n & 15; return u * 24 + w; }
  if (n < 576) { int m = n - 384, t = m >> 4, k = m & 15;
                 int w = 2 * t + (k >= 8 ? 1 : 0), u = k & 7; return 384 + u * 24 + w; }
  if (n < 704) { int m = n - 576, w = m >> 4, u = m & 15; return 576 + u * 8 + w; }
  { int m = n - 704, t = m >> 4, k = m & 15;
    int w = 2 * t + (k >= 8 ? 1 : 0), u = k & 7; return 704 + u * 8 + w; }
}

// ---------------------------------------------------------------------------
__global__ void zero_kernel(float* __restrict__ p, int n) {
  int i = blockIdx.x * blockDim.x + threadIdx.x;
  if (i < n) p[i] = 0.0f;
}

// Transpose weights to column-major bf16 (W3 also column-permuted).
__global__ void prep_weights_kernel(const float* __restrict__ W1, const float* __restrict__ W2,
                                    const float* __restrict__ W3, const float* __restrict__ b3,
                                    bf16_t* __restrict__ W1t, bf16_t* __restrict__ W2t,
                                    bf16_t* __restrict__ W3t, float* __restrict__ b3p) {
  int i = blockIdx.x * blockDim.x + threadIdx.x;
  if (i < 64 * 8)   { int n = i >> 3, k = i & 7;  W1t[i] = (bf16_t)W1[k * 64 + n]; }
  if (i < 64 * 64)  { int n = i >> 6, k = i & 63; W2t[i] = (bf16_t)W2[k * 64 + n]; }
  if (i < 768 * 64) { int n = i >> 6, k = i & 63; W3t[i] = (bf16_t)W3[(size_t)k * 768 + origcol(n)]; }
  if (i < 768)      b3p[i] = b3[origcol(i)];
}

// ---------------------------------------------------------------------------
// WMMA fragment helpers (wave32, 16x16x32 bf16 shapes per CDNA5 ISA layouts).

// A fragment from LDS row-major [16][64] bf16 stage.
// lane<16 : K = {ks..ks+7, ks+16..ks+23};  lane>=16 : +8.  (ks = 32*kstep)
__device__ __forceinline__ v16bf load_a_frag(const bf16_t* h, int lane, int kstep) {
  const bf16_t* p = h + (lane & 15) * 64 + (kstep << 5) + ((lane >> 4) << 3);
  v8bf lo = *(const v8bf*)p;
  v8bf hi = *(const v8bf*)(p + 16);
  return __builtin_shufflevector(lo, hi, 0,1,2,3,4,5,6,7,8,9,10,11,12,13,14,15);
}

// B fragment from column-major bf16 weights Wt[n*64 + k].
// lane holds column (lane&15) of the chunk, K = {ks..ks+7, ks+16..ks+23} + 8*(lane>>4)  (ks = 32*kstep)
__device__ __forceinline__ v16bf load_b_frag(const bf16_t* chunk_base, int lane, int kstep) {
  const bf16_t* p = chunk_base + (lane & 15) * 64 + ((lane >> 4) << 3) + (kstep << 5);
  v8bf lo = *(const v8bf*)p;
  v8bf hi = *(const v8bf*)(p + 16);
  return __builtin_shufflevector(lo, hi, 0,1,2,3,4,5,6,7,8,9,10,11,12,13,14,15);
}

// One 16-column tp_w chunk: 2x WMMA (K=64) + bias, staged to LDS (pitch 17).
__device__ __forceinline__ void tp_chunk(const bf16_t* __restrict__ W3t,
                                         const float* __restrict__ b3p,
                                         float* tp_w, v16bf a0, v16bf a1,
                                         int chunk, int lane) {
  v16bf b0 = load_b_frag(W3t + chunk * (16 * 64), lane, 0);
  v16bf b1 = load_b_frag(W3t + chunk * (16 * 64), lane, 1);
  v8f c;
  #pragma unroll
  for (int i = 0; i < 8; i++) c[i] = 0.0f;
  c = __builtin_amdgcn_wmma_f32_16x16x32_bf16(false, a0, false, b0, (short)0, c, false, false);
  c = __builtin_amdgcn_wmma_f32_16x16x32_bf16(false, a1, false, b1, (short)0, c, false, false);
  int col   = lane & 15;
  int rbase = (lane >> 4) << 3;           // D: lanes 0-15 -> M=r, lanes 16-31 -> M=r+8
  float bias = b3p[chunk * 16 + col];
  __builtin_amdgcn_wave_barrier();
  #pragma unroll
  for (int r = 0; r < 8; r++) tp_w[(rbase + r) * 17 + col] = c[r] + bias;
  __builtin_amdgcn_wave_barrier();
}

// ---------------------------------------------------------------------------
__global__ __launch_bounds__(WAVES * 32) void
edge_kernel(const float* __restrict__ nf, const float* __restrict__ esh,
            const float* __restrict__ basis, const int* __restrict__ eidx,
            const float* __restrict__ b1, const float* __restrict__ b2,
            const bf16_t* __restrict__ W1t, const bf16_t* __restrict__ W2t,
            const bf16_t* __restrict__ W3t, const float* __restrict__ b3p,
            float* __restrict__ agg) {
  __shared__ __align__(32) bf16_t hA[WAVES][16 * 64];   // silu(h1) stage (bf16)
  __shared__ __align__(32) bf16_t hB[WAVES][16 * 64];   // silu(h2) stage (bf16)
  __shared__ __align__(32) float  tp[WAVES][16 * 17];   // one 16x16 tp_w chunk

  const int wave = threadIdx.x >> 5;
  const int lane = threadIdx.x & 31;
  const int tile = blockIdx.x * WAVES + wave;
  if (tile >= NTILE) return;                 // no block barriers -> safe

  bf16_t* hA_w = hA[wave];
  bf16_t* hB_w = hB[wave];
  float*  tp_w = tp[wave];

  const int half = lane >> 4;                // lane pair (L, L+16) shares edge L
  const int e    = tile * 16 + (lane & 15);
  const int srcn = eidx[e];
  const int dstn = eidx[N_EDGES + e];
  const float* f = nf + (size_t)srcn * DIM_IN;

  const float sh0  = esh[(size_t)e * 4 + 0];
  const float sh1x = esh[(size_t)e * 4 + 1];
  const float sh1y = esh[(size_t)e * 4 + 2];
  const float sh1z = esh[(size_t)e * 4 + 3];

  // per-lane-half scalar multipliers (u range = half*8 .. half*8+7)
  float sxm[8], am[8];
  {
    const float* fp = f + half * 8;
    #pragma unroll
    for (int k = 0; k < 8; k++) { sxm[k] = fp[k]; am[k] = sxm[k] * sh0; }
  }
  float dot_vv[8], vxs[8][3];
  #pragma unroll
  for (int u = 0; u < 8; u++) {
    float v0 = f[16 + u * 3 + 0], v1 = f[16 + u * 3 + 1], v2 = f[16 + u * 3 + 2];
    dot_vv[u] = (v0 * sh1x + v1 * sh1y + v2 * sh1z) * INV_SQRT3;
    vxs[u][0] = v0 * sh0; vxs[u][1] = v1 * sh0; vxs[u][2] = v2 * sh0;
  }
  float* aggrow = agg + (size_t)dstn * 48;

  const int col0  = lane & 15;
  const int rbase = half << 3;

  // ---- layer 1: basis(16x8, K padded to 32) @ W1 -> silu -> hA ----
  v16bf a0;
  #pragma unroll
  for (int i = 0; i < 16; i++) a0[i] = (bf16_t)0.0f;
  if (lane < 16) {
    const float* bp = basis + (size_t)e * 8;
    #pragma unroll
    for (int i = 0; i < 8; i++) a0[i] = (bf16_t)bp[i];   // K=0..7; rest zero
  }
  #pragma unroll
  for (int nc = 0; nc < 4; nc++) {
    v16bf b;
    #pragma unroll
    for (int i = 0; i < 16; i++) b[i] = (bf16_t)0.0f;
    if (lane < 16) {                                     // B rows k=0..7 only
      const bf16_t* p = W1t + (nc * 16 + lane) * 8;
      #pragma unroll
      for (int i = 0; i < 8; i++) b[i] = p[i];
    }
    v8f c;
    #pragma unroll
    for (int i = 0; i < 8; i++) c[i] = 0.0f;
    c = __builtin_amdgcn_wmma_f32_16x16x32_bf16(false, a0, false, b, (short)0, c, false, false);
    int col = nc * 16 + col0;
    float bias = b1[col];
    #pragma unroll
    for (int r = 0; r < 8; r++) hA_w[(rbase + r) * 64 + col] = (bf16_t)silu_f(c[r] + bias);
  }
  __builtin_amdgcn_wave_barrier();

  // ---- layer 2: hA(16x64) @ W2(64x64) -> silu -> hB ----
  v16bf a1k0 = load_a_frag(hA_w, lane, 0);
  v16bf a1k1 = load_a_frag(hA_w, lane, 1);
  #pragma unroll
  for (int nc = 0; nc < 4; nc++) {
    v16bf b0  = load_b_frag(W2t + nc * (16 * 64), lane, 0);
    v16bf b1f = load_b_frag(W2t + nc * (16 * 64), lane, 1);
    v8f c;
    #pragma unroll
    for (int i = 0; i < 8; i++) c[i] = 0.0f;
    c = __builtin_amdgcn_wmma_f32_16x16x32_bf16(false, a1k0, false, b0,  (short)0, c, false, false);
    c = __builtin_amdgcn_wmma_f32_16x16x32_bf16(false, a1k1, false, b1f, (short)0, c, false, false);
    int col = nc * 16 + col0;
    float bias = b2[col];
    #pragma unroll
    for (int r = 0; r < 8; r++) hB_w[(rbase + r) * 64 + col] = (bf16_t)silu_f(c[r] + bias);
  }
  __builtin_amdgcn_wave_barrier();

  v16bf a2k0 = load_a_frag(hB_w, lane, 0);
  v16bf a2k1 = load_a_frag(hB_w, lane, 1);

  // my edge's 8-column slice of each staged chunk
  const float* myrow = tp_w + (lane & 15) * 17 + (half << 3);

  // ---- section 1: chunks 0..23 (w_ss -> out_s[w]) ----
  for (int w = 0; w < 24; w++) {
    tp_chunk(W3t, b3p, tp_w, a2k0, a2k1, w, lane);
    float acc = 0.0f;
    #pragma unroll
    for (int k = 0; k < 8; k++) acc += am[k] * myrow[k];
    acc += __shfl_xor(acc, 16, 32);
    if (lane < 16) unsafeAtomicAdd(aggrow + w, acc * ALPHA);
  }
  // ---- section 2: chunks 24..35 (w_vv -> out_s), half owns w = 2t+half ----
  for (int t = 0; t < 12; t++) {
    tp_chunk(W3t, b3p, tp_w, a2k0, a2k1, 24 + t, lane);
    float acc = 0.0f;
    #pragma unroll
    for (int k = 0; k < 8; k++) acc += dot_vv[k] * myrow[k];
    unsafeAtomicAdd(aggrow + (2 * t + half), acc * ALPHA);
  }
  // ---- section 3: chunks 36..43 (w_sv -> out_v[w][i]) ----
  for (int w = 0; w < 8; w++) {
    tp_chunk(W3t, b3p, tp_w, a2k0, a2k1, 36 + w, lane);
    float acc = 0.0f;
    #pragma unroll
    for (int k = 0; k < 8; k++) acc += sxm[k] * myrow[k];
    acc += __shfl_xor(acc, 16, 32);
    if (lane < 16) {
      float t3 = acc * ALPHA;
      unsafeAtomicAdd(aggrow + 24 + w * 3 + 0, t3 * sh1x);
      unsafeAtomicAdd(aggrow + 24 + w * 3 + 1, t3 * sh1y);
      unsafeAtomicAdd(aggrow + 24 + w * 3 + 2, t3 * sh1z);
    }
  }
  // ---- section 4: chunks 44..47 (w_vs -> out_v), half owns w = 2t+half ----
  for (int t = 0; t < 4; t++) {
    tp_chunk(W3t, b3p, tp_w, a2k0, a2k1, 44 + t, lane);
    float ai0 = 0.0f, ai1 = 0.0f, ai2 = 0.0f;
    #pragma unroll
    for (int k = 0; k < 8; k++) {
      ai0 += vxs[k][0] * myrow[k];
      ai1 += vxs[k][1] * myrow[k];
      ai2 += vxs[k][2] * myrow[k];
    }
    int w = 2 * t + half;
    unsafeAtomicAdd(aggrow + 24 + w * 3 + 0, ai0 * ALPHA);
    unsafeAtomicAdd(aggrow + 24 + w * 3 + 1, ai1 * ALPHA);
    unsafeAtomicAdd(aggrow + 24 + w * 3 + 2, ai2 * ALPHA);
  }
}

// ---------------------------------------------------------------------------
__global__ __launch_bounds__(256) void node_kernel(
    const float* __restrict__ agg, const float* __restrict__ si_w0,
    const float* __restrict__ si_w1, float* __restrict__ sv, float* __restrict__ stats) {
  __shared__ float st[40];
  int tid = threadIdx.x;
  if (tid < 40) st[tid] = 0.0f;
  __syncthreads();
  int n = blockIdx.x * blockDim.x + tid;
  if (n < N_NODES) {
    const float* a = agg + (size_t)n * 48;
    float s[16], v[8][3];
    #pragma unroll
    for (int u = 0; u < 16; u++) s[u] = silu_f(a[u]);
    #pragma unroll
    for (int u = 0; u < 8; u++) {
      float g = sigmoid_f(a[16 + u]);
      v[u][0] = a[24 + u * 3 + 0] * g;
      v[u][1] = a[24 + u * 3 + 1] * g;
      v[u][2] = a[24 + u * 3 + 2] * g;
    }
    float* o = sv + (size_t)n * 40;
    #pragma unroll
    for (int w = 0; w < 16; w++) {
      float acc = 0.0f;
      #pragma unroll
      for (int u = 0; u < 16; u++) acc += s[u] * si_w0[u * 16 + w];
      acc *= 0.25f;                       // 1/sqrt(MUL0)
      o[w] = acc;
      atomicAdd(&st[w], acc);
      atomicAdd(&st[16 + w], acc * acc);
    }
    #pragma unroll
    for (int w = 0; w < 8; w++) {
      float a0 = 0.0f, a1 = 0.0f, a2 = 0.0f;
      #pragma unroll
      for (int u = 0; u < 8; u++) {
        float wt = si_w1[u * 8 + w];
        a0 += v[u][0] * wt; a1 += v[u][1] * wt; a2 += v[u][2] * wt;
      }
      a0 *= INV_SQRT8; a1 *= INV_SQRT8; a2 *= INV_SQRT8;
      o[16 + w * 3 + 0] = a0; o[16 + w * 3 + 1] = a1; o[16 + w * 3 + 2] = a2;
      atomicAdd(&st[32 + w], (a0 * a0 + a1 * a1 + a2 * a2) * (1.0f / 3.0f));
    }
  }
  __syncthreads();
  if (tid < 40) unsafeAtomicAdd(stats + tid, st[tid]);
}

// ---------------------------------------------------------------------------
__global__ __launch_bounds__(256) void finalize_kernel(
    const float* __restrict__ sv, const float* __restrict__ stats,
    const float* __restrict__ nf, const float* __restrict__ bn_ws,
    const float* __restrict__ bn_bs, const float* __restrict__ bn_wv,
    float* __restrict__ out) {
  int n = blockIdx.x * blockDim.x + threadIdx.x;
  if (n >= N_NODES) return;
  const float invN = 1.0f / (float)N_NODES;
  const float* r = sv + (size_t)n * 40;
  const float* f = nf + (size_t)n * 40;
  float* o = out + (size_t)n * 40;
  #pragma unroll
  for (int w = 0; w < 16; w++) {
    float mu  = stats[w] * invN;
    float var = stats[16 + w] * invN - mu * mu;   // E[s^2]-mu^2 == E[(s-mu)^2]
    float sc  = rsqrtf(var + 1e-5f) * bn_ws[w];
    o[w] = (r[w] - mu) * sc + bn_bs[w] + f[w];
  }
  #pragma unroll
  for (int w = 0; w < 8; w++) {
    float sc = rsqrtf(stats[32 + w] * invN + 1e-5f) * bn_wv[w];
    o[16 + w * 3 + 0] = r[16 + w * 3 + 0] * sc + f[16 + w * 3 + 0];
    o[16 + w * 3 + 1] = r[16 + w * 3 + 1] * sc + f[16 + w * 3 + 1];
    o[16 + w * 3 + 2] = r[16 + w * 3 + 2] * sc + f[16 + w * 3 + 2];
  }
}

// ---------------------------------------------------------------------------
extern "C" void kernel_launch(void* const* d_in, const int* in_sizes, int n_in,
                              void* d_out, int out_size, void* d_ws, size_t ws_size,
                              hipStream_t stream) {
  const float* nf    = (const float*)d_in[0];
  const float* esh   = (const float*)d_in[1];
  const float* basis = (const float*)d_in[2];
  const int*   eidx  = (const int*)d_in[3];
  const float* W1    = (const float*)d_in[4];
  const float* b1    = (const float*)d_in[5];
  const float* W2    = (const float*)d_in[6];
  const float* b2    = (const float*)d_in[7];
  const float* W3    = (const float*)d_in[8];
  const float* b3    = (const float*)d_in[9];
  const float* si_w0 = (const float*)d_in[10];
  const float* si_w1 = (const float*)d_in[11];
  const float* bn_ws = (const float*)d_in[12];
  const float* bn_bs = (const float*)d_in[13];
  const float* bn_wv = (const float*)d_in[14];

  float*  ws    = (float*)d_ws;               // needs ~17.8 MB of workspace
  float*  agg   = ws + OFF_AGG;
  float*  stats = ws + OFF_STATS;
  float*  sv    = ws + OFF_SV;
  float*  b3p   = ws + OFF_B3P;
  bf16_t* wbase = (bf16_t*)(ws + OFF_WBF16);
  bf16_t* W1t   = wbase;                      // 512 bf16
  bf16_t* W2t   = wbase + 512;                // 4096 bf16
  bf16_t* W3t   = wbase + 512 + 4096;         // 49152 bf16 (column-permuted)

  const int zn = OFF_STATS + 64;              // zero agg + stats contiguously
  zero_kernel<<<(zn + 255) / 256, 256, 0, stream>>>(ws, zn);
  prep_weights_kernel<<<(768 * 64 + 255) / 256, 256, 0, stream>>>(
      W1, W2, W3, b3, W1t, W2t, W3t, b3p);
  edge_kernel<<<(NTILE + WAVES - 1) / WAVES, WAVES * 32, 0, stream>>>(
      nf, esh, basis, eidx, b1, b2, W1t, W2t, W3t, b3p, agg);
  node_kernel<<<(N_NODES + 255) / 256, 256, 0, stream>>>(agg, si_w0, si_w1, sv, stats);
  finalize_kernel<<<(N_NODES + 255) / 256, 256, 0, stream>>>(
      sv, stats, nf, bn_ws, bn_bs, bn_wv, (float*)d_out);
}
